// CausalI2VCrossAttention_67405216743832
// MI455X (gfx1250) — matrix operand order repaired and checked by reading, not compiled, not run on hardware
//
#include <hip/hip_runtime.h>
#include <stdint.h>

// ---------------------------------------------------------------------------
// CDNA5 (gfx1250) CausalI2VCrossAttention.
// - v_wmma_f32_16x16x32_bf16 everywhere (fp32 accum)
// - one-time fp32->bf16 conversion (v_cvt_pk_bf16_f32) halves GEMM streams
// - GEMM staging via global_load_async_to_lds_b128 (ASYNCcnt) + double buffer
// Workspace (~203 MB): xbf 67M | ctxbf 6.3M | 6 weights 50.3M | Qbf 67M |
//                      Ktxt/Vtxt 4.2M ea | Kimg/Vimg 2.1M ea ; attn reuses xbf
// ---------------------------------------------------------------------------

#define DIMC 2048
#define HD   128

typedef __attribute__((ext_vector_type(16))) __bf16 v16bf;
typedef __attribute__((ext_vector_type(8)))  float  v8f;

struct __align__(16) U4 { unsigned int x, y, z, w; };
union FragAB { v16bf v; U4 q[2]; };

// Pack two f32 -> two bf16 in one dword (one VALU op).
__device__ __forceinline__ unsigned int pk2(float lo, float hi) {
  unsigned int r;
  asm("v_cvt_pk_bf16_f32 %0, %1, %2" : "=v"(r) : "v"(lo), "v"(hi));
  return r;
}
__device__ __forceinline__ unsigned short f2bf(float f) {
  return (unsigned short)pk2(f, f);
}
__device__ __forceinline__ float bf2f(unsigned int u) {      // u = bf16 bits
  return __builtin_bit_cast(float, u << 16);
}

__device__ __forceinline__ v8f zero8() {
  v8f z;
  #pragma unroll
  for (int i = 0; i < 8; ++i) z[i] = 0.0f;
  return z;
}

// Async global->LDS copy (16B per lane), tracked by ASYNCcnt.
__device__ __forceinline__ void async_ld128(unsigned lds_off, const void* gaddr) {
  asm volatile("global_load_async_to_lds_b128 %0, %1, off"
               :: "v"(lds_off), "v"(gaddr) : "memory");
}
__device__ __forceinline__ void wait_async0() {
  asm volatile("s_wait_asynccnt 0x0" ::: "memory");
}

// ---------------------------------------------------------------------------
// Bulk fp32 -> bf16 conversion (8 elements / thread).  n8 = n/8.
// ---------------------------------------------------------------------------
__global__ __launch_bounds__(256)
void cvt_f32_bf16(const float* __restrict__ in, unsigned short* __restrict__ out,
                  long long n8)
{
  long long i = (long long)blockIdx.x * 256 + threadIdx.x;
  if (i >= n8) return;
  const float* p = in + i * 8;
  float4 a = *(const float4*)p;
  float4 b = *(const float4*)(p + 4);
  uint4 o = make_uint4(pk2(a.x, a.y), pk2(a.z, a.w), pk2(b.x, b.y), pk2(b.z, b.w));
  *(uint4*)(out + i * 8) = o;
}

// ---------------------------------------------------------------------------
// GEMM: C[M,N] = A[M,K](bf16) * W[N,K](bf16)^T + bias
// A rows chunked: row gr -> A + (gr/Lsub)*strideA + (gr%Lsub)*K (clamped).
// Double-buffered LDS, staged with async global->LDS b128 copies.
// Block 256 thr (8 waves); tile 128x128x32; wave tile 32x64 (2x4 frags).
// ---------------------------------------------------------------------------
#define BM 128
#define BN 128
#define BK 32
#define LDP 40   // padded LDS row (ushorts): 80B, 16B multiple

__global__ __launch_bounds__(256)
void wmma_gemm_bf16(const unsigned short* __restrict__ A, int Lsub, long long strideA,
                    const unsigned short* __restrict__ W, const float* __restrict__ bias,
                    float* __restrict__ Cf, unsigned short* __restrict__ Cbf,
                    int M, int N, int K)
{
  __shared__ __align__(16) unsigned short As[2][BM * LDP];
  __shared__ __align__(16) unsigned short Bs[2][BN * LDP];

  const int tid  = threadIdx.x;
  const int lane = tid & 31;
  const int wave = tid >> 5;
  const int wm = wave >> 1, wn = wave & 1;
  const int l15 = lane & 15, hs = lane >> 4;
  const long long tileM = (long long)blockIdx.y * BM;
  const long long tileN = (long long)blockIdx.x * BN;

  // LDS byte offsets (addrspacecast(3->0) keeps the LDS offset in low 32 bits)
  const unsigned asBase = (unsigned)(uintptr_t)&As[0][0];
  const unsigned bsBase = (unsigned)(uintptr_t)&Bs[0][0];
  const unsigned bufBytes = BM * LDP * 2;

  // Per-thread staging: 512 16B-chunks per matrix tile / 256 threads = 2 each
  const unsigned short* aPtr[2]; const unsigned short* wPtr[2];
  unsigned aLds[2], bLds[2];
  #pragma unroll
  for (int p = 0; p < 2; ++p) {
    int i = tid + p * 256;          // 128 rows x 4 chunks (8 bf16 each)
    int row = i >> 2, ck = (i & 3) * 8;
    long long gr = tileM + row;
    if (gr >= M) gr = M - 1;        // clamp; C rows >= M masked at store
    long long chk = gr / Lsub, rem = gr - chk * (long long)Lsub;
    aPtr[p] = A + chk * strideA + rem * (long long)K + ck;
    wPtr[p] = W + (tileN + row) * (long long)K + ck;
    aLds[p] = asBase + (unsigned)(row * LDP + ck) * 2;
    bLds[p] = bsBase + (unsigned)(row * LDP + ck) * 2;
  }

  v8f acc[2][4];
  #pragma unroll
  for (int m = 0; m < 2; ++m)
    #pragma unroll
    for (int n = 0; n < 4; ++n) acc[m][n] = zero8();

  // Prologue: async-stage tile 0 into buffer 0
  #pragma unroll
  for (int p = 0; p < 2; ++p) {
    async_ld128(aLds[p], aPtr[p]);
    async_ld128(bLds[p], wPtr[p]);
  }

  unsigned cur = 0;
  for (int kt = 0; kt < K; kt += BK) {
    wait_async0();        // my async copies landed in LDS
    __syncthreads();      // everyone's landed; previous reads all consumed

    if (kt + BK < K) {    // async-stage next tile into the other buffer
      unsigned nb = (cur ^ 1u) * bufBytes;
      #pragma unroll
      for (int p = 0; p < 2; ++p) {
        async_ld128(aLds[p] + nb, aPtr[p] + kt + BK);
        async_ld128(bLds[p] + nb, wPtr[p] + kt + BK);
      }
    }

    const unsigned short* AsC = &As[cur][0];
    const unsigned short* BsC = &Bs[cur][0];

    // A frag (16x32): lane row = l15; K = hs*8+{0..7} and 16+hs*8+{0..7}
    FragAB a[2], b[4];
    #pragma unroll
    for (int m = 0; m < 2; ++m) {
      const unsigned short* s = AsC + (wm * 32 + m * 16 + l15) * LDP + hs * 8;
      a[m].q[0] = *(const U4*)(s);
      a[m].q[1] = *(const U4*)(s + 16);
    }
    // B frag (32x16): lane col = l15; K = hs*8+{0..7} and 16+hs*8+{0..7}
    #pragma unroll
    for (int n = 0; n < 4; ++n) {
      const unsigned short* s = BsC + (wn * 64 + n * 16 + l15) * LDP + hs * 8;
      b[n].q[0] = *(const U4*)(s);
      b[n].q[1] = *(const U4*)(s + 16);
    }
    #pragma unroll
    for (int m = 0; m < 2; ++m)
      #pragma unroll
      for (int n = 0; n < 4; ++n)
        acc[m][n] = __builtin_amdgcn_wmma_f32_16x16x32_bf16(
            false, a[m].v, false, b[n].v, (short)0, acc[m][n], false, false);
    cur ^= 1u;
  }

  // Epilogue. D layout: VGPR r -> row = r + 8*hs, col = l15.
  #pragma unroll
  for (int m = 0; m < 2; ++m) {
    long long rowBase = tileM + wm * 32 + m * 16 + hs * 8;
    #pragma unroll
    for (int n = 0; n < 4; ++n) {
      long long col = tileN + wn * 64 + n * 16 + l15;
      float bv = bias[col];
      #pragma unroll
      for (int r = 0; r < 8; ++r) {
        long long row = rowBase + r;
        if (row < M) {
          float v = acc[m][n][r] + bv;
          if (Cf) Cf[row * (long long)N + col] = v;
          else    Cbf[row * (long long)N + col] = f2bf(v);
        }
      }
    }
  }
}

// ---------------------------------------------------------------------------
// In-place row-wise RMSNorm over DIM=2048, bf16 data.  One block per row.
// ---------------------------------------------------------------------------
__global__ __launch_bounds__(256)
void rmsnorm_bf16(unsigned short* __restrict__ d, const float* __restrict__ w)
{
  __shared__ float red[256];
  const long long row = blockIdx.x;
  unsigned short* x = d + row * (long long)DIMC;
  const int base = threadIdx.x * 8;
  uint4 raw = *(const uint4*)(x + base);
  unsigned u[4] = {raw.x, raw.y, raw.z, raw.w};
  float v[8];
  #pragma unroll
  for (int i = 0; i < 4; ++i) {
    v[2 * i]     = bf2f(u[i] & 0xffffu);
    v[2 * i + 1] = __builtin_bit_cast(float, u[i] & 0xffff0000u);
  }
  float ss = 0.f;
  #pragma unroll
  for (int i = 0; i < 8; ++i) ss += v[i] * v[i];
  red[threadIdx.x] = ss;
  __syncthreads();
  for (int s = 128; s > 0; s >>= 1) {
    if ((int)threadIdx.x < s) red[threadIdx.x] += red[threadIdx.x + s];
    __syncthreads();
  }
  float scale = rsqrtf(red[0] * (1.0f / DIMC) + 1e-6f);
  float4 w0 = *(const float4*)(w + base);
  float4 w1 = *(const float4*)(w + base + 4);
  uint4 o = make_uint4(pk2(v[0] * scale * w0.x, v[1] * scale * w0.y),
                       pk2(v[2] * scale * w0.z, v[3] * scale * w0.w),
                       pk2(v[4] * scale * w1.x, v[5] * scale * w1.y),
                       pk2(v[6] * scale * w1.z, v[7] * scale * w1.w));
  *(uint4*)(x + base) = o;
}

// ---------------------------------------------------------------------------
// Flash attention, two independent softmax sources (img, txt) summed.
// Grid: (L/128, NH, B). Block 256 thr; each wave owns 16 queries.
// Output written as bf16 (feeds the bf16 O-projection GEMM).
// ---------------------------------------------------------------------------
#define KT   32
#define KLDP 136   // 128+8 ushorts -> 272B rows (16B multiple)
#define VLDP 48    // 32+16 ushorts -> 96B rows
#define PLDP 40    // 32+8  ushorts -> 80B rows

__global__ __launch_bounds__(256)
void wmma_flash_attn(const unsigned short* __restrict__ Q,
                     const unsigned short* __restrict__ Kt,
                     const unsigned short* __restrict__ Vt,
                     const unsigned short* __restrict__ Ki,
                     const unsigned short* __restrict__ Vi,
                     unsigned short* __restrict__ Out, int L, int LkT, int LkI)
{
  __shared__ __align__(16) unsigned short Ks[KT * KLDP];
  __shared__ __align__(16) unsigned short Vs[HD * VLDP];
  __shared__ __align__(16) unsigned short Ps[8 * 16 * PLDP];

  const int tid = threadIdx.x, lane = tid & 31, wave = tid >> 5;
  const int l15 = lane & 15, hs = lane >> 4;
  const int h = blockIdx.y, b = blockIdx.z;
  const int q0 = blockIdx.x * 128 + wave * 16;

  // Q fragments: A-layout, 16 queries x 128 dims = 4 K-chunks of 32
  const unsigned short* qb = Q + ((long long)b * L + q0) * DIMC + h * HD;
  FragAB qf[4];
  #pragma unroll
  for (int kk = 0; kk < 4; ++kk) {
    const unsigned short* s = qb + (long long)l15 * DIMC + kk * 32 + hs * 8;
    qf[kk].q[0] = *(const U4*)(s);
    qf[kk].q[1] = *(const U4*)(s + 16);
  }

  const float sc = 0.08838834764831845f;   // 1/sqrt(128)
  v8f osum[8];
  #pragma unroll
  for (int f = 0; f < 8; ++f) osum[f] = zero8();

  for (int phase = 0; phase < 2; ++phase) {
    const unsigned short* Kp = phase ? Kt : Ki;
    const unsigned short* Vp = phase ? Vt : Vi;
    const int Lk = phase ? LkT : LkI;
    const unsigned short* kb = Kp + (long long)b * Lk * DIMC + h * HD;
    const unsigned short* vb = Vp + (long long)b * Lk * DIMC + h * HD;

    float mrow[8], lrow[8];
    v8f of[8];
    #pragma unroll
    for (int r = 0; r < 8; ++r) { mrow[r] = -1e30f; lrow[r] = 0.f; }
    #pragma unroll
    for (int f = 0; f < 8; ++f) of[f] = zero8();

    const int ntiles = (Lk + KT - 1) / KT;
    for (int t = 0; t < ntiles; ++t) {
      // Cooperative staging: K rows; V transposed (Vs[d][key])
      #pragma unroll
      for (int p = 0; p < 2; ++p) {
        int i = tid * 2 + p;            // 512 uint4 slots: 32 keys x 16 chunks
        int key = i >> 4;
        int c   = i & 15;
        int gk  = t * KT + key;
        if (gk >= Lk) gk = Lk - 1;      // clamp (masked downstream)
        uint4 kv = *(const uint4*)(kb + (long long)gk * DIMC + c * 8);
        uint4 vv = *(const uint4*)(vb + (long long)gk * DIMC + c * 8);
        *(uint4*)&Ks[key * KLDP + c * 8] = kv;
        const unsigned short* ve = (const unsigned short*)&vv;
        #pragma unroll
        for (int e = 0; e < 8; ++e) Vs[(c * 8 + e) * VLDP + key] = ve[e];
      }
      __syncthreads();

      // Scores S = Q * K^T  (two 16-key halves of the 32-key tile)
      v8f Sa = zero8(), Sb = zero8();
      #pragma unroll
      for (int kk = 0; kk < 4; ++kk) {
        FragAB ba, bb;
        const unsigned short* s0 = &Ks[l15 * KLDP + kk * 32 + hs * 8];
        ba.q[0] = *(const U4*)(s0); ba.q[1] = *(const U4*)(s0 + 16);
        const unsigned short* s1 = &Ks[(16 + l15) * KLDP + kk * 32 + hs * 8];
        bb.q[0] = *(const U4*)(s1); bb.q[1] = *(const U4*)(s1 + 16);
        Sa = __builtin_amdgcn_wmma_f32_16x16x32_bf16(false, qf[kk].v, false, ba.v, (short)0, Sa, false, false);
        Sb = __builtin_amdgcn_wmma_f32_16x16x32_bf16(false, qf[kk].v, false, bb.v, (short)0, Sb, false, false);
      }

      // Online softmax (xor-shuffles 1/2/4/8 reduce within each 16-lane half)
      const int keyA = t * KT + l15;
      const int keyB = keyA + 16;
      float sA[8], sB[8], mt[8], corr[8];
      #pragma unroll
      for (int r = 0; r < 8; ++r) {
        sA[r] = (keyA < Lk) ? Sa[r] * sc : -1e30f;
        sB[r] = (keyB < Lk) ? Sb[r] * sc : -1e30f;
        float v = fmaxf(sA[r], sB[r]);
        v = fmaxf(v, __shfl_xor(v, 1));
        v = fmaxf(v, __shfl_xor(v, 2));
        v = fmaxf(v, __shfl_xor(v, 4));
        v = fmaxf(v, __shfl_xor(v, 8));
        mt[r] = v;
      }
      #pragma unroll
      for (int r = 0; r < 8; ++r) {
        float nm = fmaxf(mrow[r], mt[r]);
        corr[r] = __expf(mrow[r] - nm);
        mrow[r] = nm;
        float pa = __expf(sA[r] - nm);
        float pb = __expf(sB[r] - nm);
        sA[r] = pa; sB[r] = pb;
        float s = pa + pb;
        s += __shfl_xor(s, 1);
        s += __shfl_xor(s, 2);
        s += __shfl_xor(s, 4);
        s += __shfl_xor(s, 8);
        lrow[r] = lrow[r] * corr[r] + s;
      }
      #pragma unroll
      for (int f = 0; f < 8; ++f)
        #pragma unroll
        for (int r = 0; r < 8; ++r) of[f][r] *= corr[r];

      // Re-stage P (D-layout -> A-layout) through per-wave LDS as bf16.
      unsigned short* pw = &Ps[(wave * 16 + hs * 8) * PLDP];
      #pragma unroll
      for (int r = 0; r < 8; ++r) {
        pw[r * PLDP + l15]      = f2bf(sA[r]);
        pw[r * PLDP + 16 + l15] = f2bf(sB[r]);
      }
      asm volatile("" ::: "memory");   // keep DS store->load program order (same wave)
      FragAB pf;
      const unsigned short* pr = &Ps[(wave * 16 + l15) * PLDP + hs * 8];
      pf.q[0] = *(const U4*)(pr);
      pf.q[1] = *(const U4*)(pr + 16);

      // O += P * V  (V B-frags from transposed LDS tile)
      #pragma unroll
      for (int f = 0; f < 8; ++f) {
        FragAB vf;
        const unsigned short* vsrc = &Vs[(f * 16 + l15) * VLDP + hs * 8];
        vf.q[0] = *(const U4*)(vsrc); vf.q[1] = *(const U4*)(vsrc + 16);
        of[f] = __builtin_amdgcn_wmma_f32_16x16x32_bf16(false, pf.v, false, vf.v, (short)0, of[f], false, false);
      }
      __syncthreads();
    }
    #pragma unroll
    for (int f = 0; f < 8; ++f)
      #pragma unroll
      for (int r = 0; r < 8; ++r) osum[f][r] += of[f][r] / lrow[r];
  }

  unsigned short* ob = Out + ((long long)b * L + q0) * DIMC + h * HD;
  #pragma unroll
  for (int f = 0; f < 8; ++f)
    #pragma unroll
    for (int r = 0; r < 8; ++r)
      ob[(long long)(hs * 8 + r) * DIMC + f * 16 + l15] = f2bf(osum[f][r]);
}

// ---------------------------------------------------------------------------
extern "C" void kernel_launch(void* const* d_in, const int* in_sizes, int n_in,
                              void* d_out, int out_size, void* d_ws, size_t ws_size,
                              hipStream_t stream) {
  (void)in_sizes; (void)n_in; (void)out_size; (void)ws_size;

  const float* x        = (const float*)d_in[0];
  const float* ctx      = (const float*)d_in[1];
  const float* wsrc[6]  = { (const float*)d_in[2],  (const float*)d_in[4],
                            (const float*)d_in[6],  (const float*)d_in[8],
                            (const float*)d_in[10], (const float*)d_in[12] };
  const float* q_b      = (const float*)d_in[3];
  const float* k_b      = (const float*)d_in[5];
  const float* v_b      = (const float*)d_in[7];
  const float* k_img_b  = (const float*)d_in[9];
  const float* v_img_b  = (const float*)d_in[11];
  const float* o_b      = (const float*)d_in[13];
  const float* norm_q   = (const float*)d_in[14];
  const float* norm_k   = (const float*)d_in[15];
  const float* norm_ki  = (const float*)d_in[16];

  const int B = 2, L = 8192, Ltxt = 512, Limg = 257, Lctx = 769;
  const int Mq = B * L;                // 16384
  const int Mt = B * Ltxt;             // 1024
  const int Mi = B * Limg;             // 514
  const long long ctxStride = (long long)Lctx * DIMC;

  typedef unsigned short u16;
  char* ws = (char*)d_ws;
  size_t off = 0;
  u16* xbf   = (u16*)(ws + off); off += (size_t)Mq * DIMC * 2;
  u16* ctxbf = (u16*)(ws + off); off += (size_t)B * Lctx * DIMC * 2;
  u16* wbf[6];
  for (int i = 0; i < 6; ++i) { wbf[i] = (u16*)(ws + off); off += (size_t)DIMC * DIMC * 2; }
  u16* Qbf  = (u16*)(ws + off); off += (size_t)Mq * DIMC * 2;
  u16* Ktxt = (u16*)(ws + off); off += (size_t)Mt * DIMC * 2;
  u16* Vtxt = (u16*)(ws + off); off += (size_t)Mt * DIMC * 2;
  u16* Kimg = (u16*)(ws + off); off += (size_t)Mi * DIMC * 2;
  u16* Vimg = (u16*)(ws + off); off += (size_t)Mi * DIMC * 2;
  u16* attnbf = xbf;                   // xbf is dead after the projections

  const dim3 blk(256);
  auto gg  = [](int M) { return dim3(DIMC / BN, (M + BM - 1) / BM); };
  auto cgr = [](long long n8) { return dim3((unsigned)((n8 + 255) / 256)); };

  // One-time fp32 -> bf16 conversions
  long long n8x = (long long)Mq * DIMC / 8;
  long long n8c = (long long)B * Lctx * DIMC / 8;
  long long n8w = (long long)DIMC * DIMC / 8;
  cvt_f32_bf16<<<cgr(n8x), blk, 0, stream>>>(x,   xbf,   n8x);
  cvt_f32_bf16<<<cgr(n8c), blk, 0, stream>>>(ctx, ctxbf, n8c);
  for (int i = 0; i < 6; ++i)
    cvt_f32_bf16<<<cgr(n8w), blk, 0, stream>>>(wsrc[i], wbf[i], n8w);

  const u16* ctxTxt = ctxbf + (long long)Limg * DIMC;

  // Q = rmsnorm(x @ q_w^T + q_b)            (in-place norm on bf16)
  wmma_gemm_bf16<<<gg(Mq), blk, 0, stream>>>(xbf, Mq, 0, wbf[0], q_b, nullptr, Qbf, Mq, DIMC, DIMC);
  rmsnorm_bf16<<<dim3(Mq), blk, 0, stream>>>(Qbf, norm_q);

  // K_txt / V_txt from context[:, 257:]
  wmma_gemm_bf16<<<gg(Mt), blk, 0, stream>>>(ctxTxt, Ltxt, ctxStride, wbf[1], k_b, nullptr, Ktxt, Mt, DIMC, DIMC);
  rmsnorm_bf16<<<dim3(Mt), blk, 0, stream>>>(Ktxt, norm_k);
  wmma_gemm_bf16<<<gg(Mt), blk, 0, stream>>>(ctxTxt, Ltxt, ctxStride, wbf[2], v_b, nullptr, Vtxt, Mt, DIMC, DIMC);

  // K_img / V_img from context[:, :257]
  wmma_gemm_bf16<<<gg(Mi), blk, 0, stream>>>(ctxbf, Limg, ctxStride, wbf[3], k_img_b, nullptr, Kimg, Mi, DIMC, DIMC);
  rmsnorm_bf16<<<dim3(Mi), blk, 0, stream>>>(Kimg, norm_ki);
  wmma_gemm_bf16<<<gg(Mi), blk, 0, stream>>>(ctxbf, Limg, ctxStride, wbf[4], v_img_b, nullptr, Vimg, Mi, DIMC, DIMC);

  // Attention (img softmax + txt softmax, summed) -> attnbf (bf16, reuses xbf)
  wmma_flash_attn<<<dim3(L / 128, 16, B), blk, 0, stream>>>(
      Qbf, Ktxt, Vtxt, Kimg, Vimg, attnbf, L, Ltxt, Limg);

  // Output projection -> d_out (fp32)
  wmma_gemm_bf16<<<gg(Mq), blk, 0, stream>>>(attnbf, Mq, 0, wbf[5], o_b, (float*)d_out, nullptr, Mq, DIMC, DIMC);
}
